// BiMPM_33122787787515
// MI455X (gfx1250) — compile-verified, matrix operand order checked by reading, not run on hardware
//
#include <hip/hip_runtime.h>

#define EPSF 1e-8f

typedef __attribute__((ext_vector_type(16))) _Float16 v16h;
typedef __attribute__((ext_vector_type(8)))  float    v8f;

__device__ __forceinline__ float sigm(float x) { return 1.0f / (1.0f + __expf(-x)); }

// ---------------------------------------------------------------------------
// WMMA helpers (CDNA5 gfx1250, wave32). D = A(16x32 f16) x B(32x16 f16) + C(f32)
// Fragment layouts per CDNA5 ISA 7.12.2.
// ---------------------------------------------------------------------------
__device__ __forceinline__ v8f wmma16(v16h a, v16h b, v8f c) {
  return __builtin_amdgcn_wmma_f32_16x16x32_f16(false, a, false, b, (short)0, c,
                                                false, false);
}

// Load a 16x32 fragment from a row-major [16+ rows][stride] f16 matrix.
// A-matrix: lane holds row (lane%16); halves of wave split K: lanes 0-15 get
// K=0..7,16..23; lanes 16-31 get K=8..15,24..31. B (as NxK rows) uses the
// symmetric layout, so loading W[N][K] rows here computes X @ W^T.
__device__ __forceinline__ v16h ld_frag(const _Float16* base, int stride) {
  const int lane = threadIdx.x & 31;
  const int row  = lane & 15;
  const int hi   = lane >> 4;
  const _Float16* r = base + (size_t)row * stride + hi * 8;
  union { v16h v; _Float16 e[16]; } f;
#pragma unroll
  for (int c = 0; c < 8; ++c) f.e[c] = r[c];
#pragma unroll
  for (int c = 0; c < 8; ++c) f.e[8 + c] = r[16 + c];
  return f.v;
}

// Same, but element k scaled by w2[k0+k] (folds perspective weight^2 into A).
__device__ __forceinline__ v16h ld_frag_scaled(const _Float16* base, int stride,
                                               const _Float16* w2, int k0) {
  const int lane = threadIdx.x & 31;
  const int row  = lane & 15;
  const int hi   = lane >> 4;
  const _Float16* r = base + (size_t)row * stride + k0 + hi * 8;
  const _Float16* s = w2 + k0 + hi * 8;
  union { v16h v; _Float16 e[16]; } f;
#pragma unroll
  for (int c = 0; c < 8; ++c) f.e[c] = (_Float16)(r[c] * s[c]);
#pragma unroll
  for (int c = 0; c < 8; ++c) f.e[8 + c] = (_Float16)(r[16 + c] * s[16 + c]);
  return f.v;
}

// C/D layout: VGPR r holds M = 8*(lane/16)+r, N = lane%16.
__device__ __forceinline__ void st_frag_f32(float* base, int stride, v8f acc) {
  const int lane = threadIdx.x & 31;
  const int col  = lane & 15;
  const int hi   = lane >> 4;
  union { v8f v; float e[8]; } f; f.v = acc;
#pragma unroll
  for (int rr = 0; rr < 8; ++rr)
    base[(size_t)(hi * 8 + rr) * stride + col] = f.e[rr];
}

__device__ __forceinline__ void st_frag_f16(_Float16* base, int stride, v8f acc) {
  const int lane = threadIdx.x & 31;
  const int col  = lane & 15;
  const int hi   = lane >> 4;
  union { v8f v; float e[8]; } f; f.v = acc;
#pragma unroll
  for (int rr = 0; rr < 8; ++rr)
    base[(size_t)(hi * 8 + rr) * stride + col] = (_Float16)f.e[rr];
}

// Async global -> LDS 16B copy (CDNA5 GLOBAL_LOAD_ASYNC_TO_LDS_B128, ASYNCcnt).
// ldsoff: byte offset into the workgroup's LDS allocation (extern smem base=0).
__device__ __forceinline__ void async_g2l_b128(unsigned int ldsoff,
                                               const void* gsrc) {
  unsigned long long ga = (unsigned long long)(uintptr_t)gsrc;
  asm volatile("global_load_async_to_lds_b128 %0, %1, off"
               :: "v"(ldsoff), "v"(ga)
               : "memory");
}

__device__ __forceinline__ void wait_asynccnt0() {
  asm volatile("s_wait_asynccnt 0x0" ::: "memory");
}

// ---------------------------------------------------------------------------
// K0: f32 -> f16 conversion
// ---------------------------------------------------------------------------
__global__ void f32_to_f16_kernel(const float* __restrict__ src,
                                  _Float16* __restrict__ dst, int n) {
  int i = blockIdx.x * 256 + threadIdx.x;
  if (i < n) dst[i] = (_Float16)src[i];
}

// ---------------------------------------------------------------------------
// K1: char LSTM. 8192 sequences (4096 p + 4096 h), T=16, E=64, H=128.
// One WG of 128 threads (4 waves) per 16 sequences. Weights resident in LDS.
// Embedding rows gathered straight into LDS with async b128 copies.
// ---------------------------------------------------------------------------
#define SMEM_CHAR ((512*64 + 512*128 + 16*64 + 16*128) * 2 + (16*128 + 16*512) * 4)
#define SX_CHAR_OFF ((512*64 + 512*128) * 2)  // LDS byte offset of sX

__global__ __launch_bounds__(128) void char_lstm_kernel(
    const int* __restrict__ p_char, const int* __restrict__ h_char,
    const _Float16* __restrict__ emb16,
    const _Float16* __restrict__ wih, const _Float16* __restrict__ whh,
    const float* __restrict__ bias, _Float16* __restrict__ seq_f16) {
  extern __shared__ char smem[];
  _Float16* sWih = (_Float16*)smem;            // [512][64]
  _Float16* sWhh = sWih + 512 * 64;            // [512][128]
  _Float16* sX   = sWhh + 512 * 128;           // [16][64]   (LDS off SX_CHAR_OFF)
  _Float16* sH   = sX + 16 * 64;               // [16][128]
  float*    sC   = (float*)(sH + 16 * 128);    // [16][128]
  float*    sZ   = sC + 16 * 128;              // [16][512]

  const int tid  = threadIdx.x;
  const int wave = tid >> 5;

  for (int i = tid; i < 512 * 64;  i += 128) sWih[i] = wih[i];
  for (int i = tid; i < 512 * 128; i += 128) sWhh[i] = whh[i];
  for (int i = tid; i < 16 * 128;  i += 128) { sH[i] = (_Float16)0.0f; sC[i] = 0.0f; }
  __syncthreads();

  const int seq0 = blockIdx.x * 16;  // 0..8191
  const int* chars = (seq0 < 4096) ? (p_char + (size_t)seq0 * 16)
                                   : (h_char + (size_t)(seq0 - 4096) * 16);

  for (int t = 0; t < 16; ++t) {
    // Gather 16 embedding rows (128 B each = 8 chunks) async into sX.
    {
      int s = tid >> 3, part = tid & 7;  // 128 threads = 16 rows x 8 chunks
      int ch = chars[s * 16 + t];
      async_g2l_b128(SX_CHAR_OFF + (unsigned)(s * 128 + part * 16),
                     emb16 + (size_t)ch * 64 + part * 8);
      wait_asynccnt0();
    }
    __syncthreads();

    // 32 N-tiles over 4 waves -> 8 each; K = 64 (input) + 128 (recurrent)
    for (int n = 0; n < 8; ++n) {
      int nt = wave * 8 + n;
      v8f acc = {};
#pragma unroll
      for (int kt = 0; kt < 2; ++kt)
        acc = wmma16(ld_frag(sX + kt * 32, 64),
                     ld_frag(sWih + (size_t)nt * 16 * 64 + kt * 32, 64), acc);
#pragma unroll
      for (int kt = 0; kt < 4; ++kt)
        acc = wmma16(ld_frag(sH + kt * 32, 128),
                     ld_frag(sWhh + (size_t)nt * 16 * 128 + kt * 32, 128), acc);
      st_frag_f32(sZ + nt * 16, 512, acc);
    }
    __syncthreads();

    for (int i = tid; i < 16 * 128; i += 128) {
      int s = i >> 7, u = i & 127;
      float zi = sZ[s * 512 + u]       + bias[u];
      float zf = sZ[s * 512 + 128 + u] + bias[128 + u];
      float zg = sZ[s * 512 + 256 + u] + bias[256 + u];
      float zo = sZ[s * 512 + 384 + u] + bias[384 + u];
      float c = sigm(zf) * sC[i] + sigm(zi) * tanhf(zg);
      float h = sigm(zo) * tanhf(c);
      sC[i] = c;
      sH[i] = (_Float16)h;
    }
    __syncthreads();
  }

  for (int i = tid; i < 16 * 128; i += 128) {
    int s = i >> 7, u = i & 127;
    seq_f16[(size_t)(seq0 + s) * 128 + u] = sH[i];
  }
}

// ---------------------------------------------------------------------------
// K2: context LSTMs. 4 persistent WGs (p-fw, p-bw, h-fw, h-bw), B=64, L=64,
// K=128 -> H=256 (gates 1024). Weights streamed from L2 (f16), states in LDS.
// X staged per step via async global->LDS b128 copies (sX is at LDS offset 0).
// ---------------------------------------------------------------------------
#define NCTX ((size_t)64 * 64 * 256)
#define SMEM_CTX ((64*128 + 64*256 + 64*1024) * 2 + (64*256) * 4)

__global__ __launch_bounds__(512) void ctx_lstm_kernel(
    const _Float16* __restrict__ seq_f16,
    const _Float16* __restrict__ wih_f, const _Float16* __restrict__ whh_f,
    const float* __restrict__ b_f,
    const _Float16* __restrict__ wih_b, const _Float16* __restrict__ whh_b,
    const float* __restrict__ b_b,
    float* __restrict__ ctx32, _Float16* __restrict__ ctx16) {
  extern __shared__ char smem[];
  _Float16* sX = (_Float16*)smem;           // [64][128]  (LDS offset 0)
  _Float16* sH = sX + 64 * 128;             // [64][256]
  _Float16* sZ = sH + 64 * 256;             // [64][1024]
  float*    sC = (float*)(sZ + 64 * 1024);  // [64][256]

  const int run  = blockIdx.x;      // 0 p-fw, 1 p-bw, 2 h-fw, 3 h-bw
  const int dir  = run & 1;
  const int side = run >> 1;
  const _Float16* wih = dir ? wih_b : wih_f;
  const _Float16* whh = dir ? whh_b : whh_f;
  const float*   bias = dir ? b_b : b_f;
  const _Float16* in  = seq_f16 + (size_t)side * 4096 * 128;
  float*    out  = ctx32 + (size_t)run * NCTX;
  _Float16* outh = ctx16 + (size_t)run * NCTX;

  const int tid  = threadIdx.x;
  const int wave = tid >> 5;  // 16 waves

  for (int i = tid; i < 64 * 256; i += 512) { sH[i] = (_Float16)0.0f; sC[i] = 0.0f; }
  __syncthreads();

  for (int t = 0; t < 64; ++t) {
    const int tt = dir ? (63 - t) : t;
    // Stage X: 64 rows x 256 B = 1024 async b128 chunks over 512 threads.
#pragma unroll
    for (int c = 0; c < 2; ++c) {
      int chunk = tid + c * 512;          // 0..1023
      int bb = chunk >> 4, part = chunk & 15;
      async_g2l_b128((unsigned)chunk * 16,
                     in + ((size_t)bb * 64 + tt) * 128 + part * 8);
    }
    wait_asynccnt0();
    __syncthreads();

    // 4 M-tiles x 64 N-tiles = 256 jobs over 16 waves
    for (int j = 0; j < 16; ++j) {
      int job = wave * 16 + j;
      int mt = job & 3, nt = job >> 2;
      if (j + 1 < 16) {  // prefetch next job's weight tiles into GL2/L0
        int nt2 = (wave * 16 + j + 1) >> 2;
        __builtin_prefetch(wih + (size_t)nt2 * 16 * 128, 0, 1);
        __builtin_prefetch(whh + (size_t)nt2 * 16 * 256, 0, 1);
      }
      v8f acc = {};
#pragma unroll
      for (int kt = 0; kt < 4; ++kt)
        acc = wmma16(ld_frag(sX + (size_t)mt * 16 * 128 + kt * 32, 128),
                     ld_frag(wih + (size_t)nt * 16 * 128 + kt * 32, 128), acc);
#pragma unroll
      for (int kt = 0; kt < 8; ++kt)
        acc = wmma16(ld_frag(sH + (size_t)mt * 16 * 256 + kt * 32, 256),
                     ld_frag(whh + (size_t)nt * 16 * 256 + kt * 32, 256), acc);
      st_frag_f16(sZ + (size_t)mt * 16 * 1024 + nt * 16, 1024, acc);
    }
    __syncthreads();

    for (int i = tid; i < 64 * 256; i += 512) {
      int bb = i >> 8, u = i & 255;
      float zi = (float)sZ[bb * 1024 + u]       + bias[u];
      float zf = (float)sZ[bb * 1024 + 256 + u] + bias[256 + u];
      float zg = (float)sZ[bb * 1024 + 512 + u] + bias[512 + u];
      float zo = (float)sZ[bb * 1024 + 768 + u] + bias[768 + u];
      float c = sigm(zf) * sC[i] + sigm(zi) * tanhf(zg);
      float h = sigm(zo) * tanhf(c);
      sC[i] = c;
      sH[i] = (_Float16)h;
      size_t o = ((size_t)bb * 64 + tt) * 256 + u;
      out[o]  = h;
      outh[o] = (_Float16)h;
    }
    __syncthreads();
  }
}

// ---------------------------------------------------------------------------
// K3: inverse norms: invn[side][dir][pp][b][l] = 1/max(||w_pp . v||, EPS)
// pp 0..3 use mp3 (fw) / mp4 (bw); pp 4 is the plain norm (attention).
// ---------------------------------------------------------------------------
__global__ __launch_bounds__(320) void norms_kernel(
    const float* __restrict__ ctx32, const float* __restrict__ mp3,
    const float* __restrict__ mp4, float* __restrict__ invn) {
  const int b    = blockIdx.x & 63;
  const int dir  = (blockIdx.x >> 6) & 1;
  const int side = blockIdx.x >> 7;
  const float* v = ctx32 + (size_t)(side * 2 + dir) * NCTX + ((size_t)b * 64) * 256;
  const float* w = dir ? mp4 : mp3;
  const int l  = threadIdx.x & 63;
  const int pp = threadIdx.x >> 6;  // 0..4
  float s = 0.0f;
  for (int k = 0; k < 256; ++k) {
    float wv = (pp < 4) ? w[pp * 256 + k] : 1.0f;
    float x  = wv * v[(size_t)l * 256 + k];
    s += x * x;
  }
  invn[((((size_t)side * 2 + dir) * 5 + pp) * 64 + b) * 64 + l] =
      1.0f / fmaxf(sqrtf(s), EPSF);
}

// ---------------------------------------------------------------------------
// K4: pairwise similarity GEMMs (maxpool-match pp 0..3, attention pp 4):
// S[dir][b][pp][i][j] = sum_k w_pp[k]^2 * p[b,i,k] * h[b,j,k]  (w=1 for pp=4)
// ---------------------------------------------------------------------------
__global__ __launch_bounds__(256) void pairwise_kernel(
    const _Float16* __restrict__ ctx16, const float* __restrict__ mp3,
    const float* __restrict__ mp4, float* __restrict__ S) {
  const int pp  = blockIdx.x % 5;
  const int b   = (blockIdx.x / 5) & 63;
  const int dir = blockIdx.x / (5 * 64);
  const _Float16* A  = ctx16 + (size_t)(0 * 2 + dir) * NCTX + ((size_t)b * 64) * 256;
  const _Float16* Bm = ctx16 + (size_t)(1 * 2 + dir) * NCTX + ((size_t)b * 64) * 256;
  const float* w = dir ? mp4 : mp3;

  __shared__ _Float16 sw2[256];
  const int tid = threadIdx.x;
  {
    float wv = (pp < 4) ? w[pp * 256 + tid] : 1.0f;
    sw2[tid] = (_Float16)(wv * wv);
  }
  __syncthreads();

  float* Sout = S + ((((size_t)dir * 64 + b) * 5 + pp) * 4096);
  const int wave = tid >> 5;  // 8 waves, 16 tile jobs
  for (int j = 0; j < 2; ++j) {
    int job = wave * 2 + j;
    int it = job >> 2, jt = job & 3;
    v8f acc = {};
#pragma unroll
    for (int kt = 0; kt < 8; ++kt)
      acc = wmma16(ld_frag_scaled(A + (size_t)it * 16 * 256, 256, sw2, kt * 32),
                   ld_frag(Bm + (size_t)jt * 16 * 256 + kt * 32, 256), acc);
    st_frag_f32(Sout + (size_t)it * 16 * 64 + jt * 16, 64, acc);
  }
}

// ---------------------------------------------------------------------------
// K5: maxpool-match reductions -> mv cols [4..8)+16*dir
// ---------------------------------------------------------------------------
__global__ __launch_bounds__(256) void maxreduce_kernel(
    const float* __restrict__ S, const float* __restrict__ invn,
    float* __restrict__ mv_p, float* __restrict__ mv_h) {
  const int b   = blockIdx.x & 63;
  const int dir = blockIdx.x >> 6;
  const int i   = threadIdx.x & 63;
  const int pp  = threadIdx.x >> 6;  // 0..3
  const float* Sb  = S + ((((size_t)dir * 64 + b) * 5 + pp) * 4096);
  const float* inp = invn + ((((size_t)0 * 2 + dir) * 5 + pp) * 64 + b) * 64;
  const float* inh = invn + ((((size_t)1 * 2 + dir) * 5 + pp) * 64 + b) * 64;

  float ai = inp[i], m = -3.4e38f;
  for (int j = 0; j < 64; ++j) m = fmaxf(m, Sb[i * 64 + j] * ai * inh[j]);
  mv_p[((size_t)b * 64 + i) * 32 + 4 + dir * 16 + pp] = m;

  float aj = inh[i], m2 = -3.4e38f;
  for (int ii = 0; ii < 64; ++ii) m2 = fmaxf(m2, Sb[ii * 64 + i] * inp[ii] * aj);
  mv_h[((size_t)b * 64 + i) * 32 + 4 + dir * 16 + pp] = m2;
}

// ---------------------------------------------------------------------------
// K6: attention features. att = S[pp=4] * invn_p[i] * invn_h[j].
// feats: [8][64*64*256]: 0..3 mean (h_fw, h_bw, p_fw, p_bw), 4..7 max (same).
// ---------------------------------------------------------------------------
__global__ __launch_bounds__(256) void attfeat_kernel(
    const float* __restrict__ S, const float* __restrict__ invn,
    const float* __restrict__ ctx32, float* __restrict__ feats) {
  const int i    = blockIdx.x & 63;
  const int b    = (blockIdx.x >> 6) & 63;
  const int dir  = (blockIdx.x >> 12) & 1;
  const int vsel = blockIdx.x >> 13;  // 0 -> v = h side, 1 -> v = p side
  const float* v = ctx32 + (size_t)((vsel ? 0 : 1) * 2 + dir) * NCTX +
                   ((size_t)b * 64) * 256;
  const float* Sb  = S + ((((size_t)dir * 64 + b) * 5 + 4) * 4096);
  const float* inp = invn + ((((size_t)0 * 2 + dir) * 5 + 4) * 64 + b) * 64;
  const float* inh = invn + ((((size_t)1 * 2 + dir) * 5 + 4) * 64 + b) * 64;

  __shared__ float satt[64];
  const int tid = threadIdx.x;
  if (tid < 64) satt[tid] = Sb[i * 64 + tid] * inp[i] * inh[tid];
  __syncthreads();

  float ssum = 0.0f;
  for (int j = 0; j < 64; ++j) ssum += satt[j];

  const int d = tid;  // 256 dims
  float mean = 0.0f, mx = -3.4e38f;
  for (int j = 0; j < 64; ++j) {
    float a  = satt[j];
    float vv = v[(size_t)j * 256 + d];
    mean += a * vv;
    mx = fmaxf(mx, a * vv);
  }
  mean = mean / fmaxf(ssum, EPSF);
  size_t base = ((size_t)b * 64 + i) * 256 + d;
  feats[(size_t)(vsel * 2 + dir) * NCTX + base]     = mean;
  feats[(size_t)(4 + vsel * 2 + dir) * NCTX + base] = mx;
}

// ---------------------------------------------------------------------------
// K7: full-match cosine. One WG per (b,l); one wave per perspective (P=4).
// cv_fixed >= 0 : compare against fixed row of Cv; < 0 : elementwise in l.
// ---------------------------------------------------------------------------
__global__ __launch_bounds__(128) void matchfull_kernel(
    const float* __restrict__ A, const float* __restrict__ Cv, int cv_fixed,
    const float* __restrict__ w, float* __restrict__ out, int col0) {
  const int b = blockIdx.x >> 6;
  const int l = blockIdx.x & 63;
  const int lane = threadIdx.x & 31;
  const int pp   = threadIdx.x >> 5;
  const float* av = A + ((size_t)b * 64 + l) * 256;
  const float* cv = Cv + ((size_t)b * 64 + (cv_fixed < 0 ? l : cv_fixed)) * 256;
  const float* wp = w + pp * 256;
  float num = 0.0f, na = 0.0f, nb = 0.0f;
  for (int k = lane; k < 256; k += 32) {
    float ww = wp[k];
    float x = ww * av[k], y = ww * cv[k];
    num += x * y; na += x * x; nb += y * y;
  }
#pragma unroll
  for (int off = 16; off > 0; off >>= 1) {
    num += __shfl_xor(num, off, 32);
    na  += __shfl_xor(na,  off, 32);
    nb  += __shfl_xor(nb,  off, 32);
  }
  if (lane == 0) {
    float d = fmaxf(sqrtf(na), EPSF) * fmaxf(sqrtf(nb), EPSF);
    out[((size_t)b * 64 + l) * 32 + col0 + pp] = num / d;
  }
}

// ---------------------------------------------------------------------------
// K8: aggregation LSTMs. 4 WGs (p-fw, p-bw, h-fw, h-bw), K=32 -> H=128.
// Both weight matrices fit in LDS.
// ---------------------------------------------------------------------------
#define SMEM_AGG ((512*32 + 512*128 + 64*32 + 64*128 + 64*512) * 2 + (64*128) * 4)

__global__ __launch_bounds__(256) void agg_lstm_kernel(
    const float* __restrict__ mv_p, const float* __restrict__ mv_h,
    const _Float16* __restrict__ wih_f, const _Float16* __restrict__ whh_f,
    const float* __restrict__ b_f,
    const _Float16* __restrict__ wih_b, const _Float16* __restrict__ whh_b,
    const float* __restrict__ b_b, float* __restrict__ xcat) {
  extern __shared__ char smem[];
  _Float16* sWih = (_Float16*)smem;            // [512][32]
  _Float16* sWhh = sWih + 512 * 32;            // [512][128]
  _Float16* sX   = sWhh + 512 * 128;           // [64][32]
  _Float16* sH   = sX + 64 * 32;               // [64][128]
  _Float16* sZ   = sH + 64 * 128;              // [64][512]
  float*    sC   = (float*)(sZ + 64 * 512);    // [64][128]

  const int run  = blockIdx.x;  // 0 p-fw, 1 p-bw, 2 h-fw, 3 h-bw
  const int dir  = run & 1;
  const int side = run >> 1;
  const float* mv = side ? mv_h : mv_p;
  const _Float16* wih = dir ? wih_b : wih_f;
  const _Float16* whh = dir ? whh_b : whh_f;
  const float*   bias = dir ? b_b : b_f;

  const int tid  = threadIdx.x;
  const int wave = tid >> 5;  // 8 waves

  for (int i = tid; i < 512 * 32;  i += 256) sWih[i] = wih[i];
  for (int i = tid; i < 512 * 128; i += 256) sWhh[i] = whh[i];
  for (int i = tid; i < 64 * 128;  i += 256) { sH[i] = (_Float16)0.0f; sC[i] = 0.0f; }
  __syncthreads();

  for (int t = 0; t < 64; ++t) {
    const int tt = dir ? (63 - t) : t;
    for (int i = tid; i < 64 * 32; i += 256) {
      int bb = i >> 5, k = i & 31;
      sX[i] = (_Float16)mv[((size_t)bb * 64 + tt) * 32 + k];
    }
    __syncthreads();

    // 4 M-tiles x 32 N-tiles = 128 jobs over 8 waves
    for (int j = 0; j < 16; ++j) {
      int job = wave * 16 + j;
      int mt = job & 3, nt = job >> 2;
      v8f acc = {};
      acc = wmma16(ld_frag(sX + (size_t)mt * 16 * 32, 32),
                   ld_frag(sWih + (size_t)nt * 16 * 32, 32), acc);
#pragma unroll
      for (int kt = 0; kt < 4; ++kt)
        acc = wmma16(ld_frag(sH + (size_t)mt * 16 * 128 + kt * 32, 128),
                     ld_frag(sWhh + (size_t)nt * 16 * 128 + kt * 32, 128), acc);
      st_frag_f16(sZ + (size_t)mt * 16 * 512 + nt * 16, 512, acc);
    }
    __syncthreads();

    for (int i = tid; i < 64 * 128; i += 256) {
      int bb = i >> 7, u = i & 127;
      float zi = (float)sZ[bb * 512 + u]       + bias[u];
      float zf = (float)sZ[bb * 512 + 128 + u] + bias[128 + u];
      float zg = (float)sZ[bb * 512 + 256 + u] + bias[256 + u];
      float zo = (float)sZ[bb * 512 + 384 + u] + bias[384 + u];
      float c = sigm(zf) * sC[i] + sigm(zi) * tanhf(zg);
      float h = sigm(zo) * tanhf(c);
      sC[i] = c;
      sH[i] = (_Float16)h;
    }
    __syncthreads();
  }

  // x = concat(agg(mv_p).hf, .hb, agg(mv_h).hf, .hb) -> column block = run*128
  for (int i = tid; i < 64 * 128; i += 256) {
    int bb = i >> 7, u = i & 127;
    xcat[(size_t)bb * 512 + run * 128 + u] = (float)sH[i];
  }
}

// ---------------------------------------------------------------------------
// K9: two linear layers (no activation between, per reference).
// ---------------------------------------------------------------------------
__global__ __launch_bounds__(256) void fc_kernel(
    const float* __restrict__ x, const float* __restrict__ fc1w,
    const float* __restrict__ fc1b, const float* __restrict__ fc2w,
    const float* __restrict__ fc2b, float* __restrict__ out) {
  const int b = blockIdx.x;
  __shared__ float y1[256];
  const int u = threadIdx.x;
  float s = fc1b[u];
  for (int k = 0; k < 512; ++k) s += x[(size_t)b * 512 + k] * fc1w[(size_t)u * 512 + k];
  y1[u] = s;
  __syncthreads();
  if (u < 3) {
    float o = fc2b[u];
    for (int k = 0; k < 256; ++k) o += y1[k] * fc2w[(size_t)u * 256 + k];
    out[(size_t)b * 3 + u] = o;
  }
}

// ---------------------------------------------------------------------------
// Host orchestration
// ---------------------------------------------------------------------------
extern "C" void kernel_launch(void* const* d_in, const int* in_sizes, int n_in,
                              void* d_out, int out_size, void* d_ws, size_t ws_size,
                              hipStream_t stream) {
  (void)in_sizes; (void)n_in; (void)out_size; (void)ws_size;
  const int*   p_char  = (const int*)d_in[0];
  const int*   h_char  = (const int*)d_in[1];
  const float* emb     = (const float*)d_in[2];
  const float* wih_c   = (const float*)d_in[3];
  const float* whh_c   = (const float*)d_in[4];
  const float* b_c     = (const float*)d_in[5];
  const float* wih_cf  = (const float*)d_in[6];
  const float* whh_cf  = (const float*)d_in[7];
  const float* b_cf    = (const float*)d_in[8];
  const float* wih_cb  = (const float*)d_in[9];
  const float* whh_cb  = (const float*)d_in[10];
  const float* b_cb    = (const float*)d_in[11];
  const float* wih_af  = (const float*)d_in[12];
  const float* whh_af  = (const float*)d_in[13];
  const float* b_af    = (const float*)d_in[14];
  const float* wih_ab  = (const float*)d_in[15];
  const float* whh_ab  = (const float*)d_in[16];
  const float* b_ab    = (const float*)d_in[17];
  const float* fc1_w   = (const float*)d_in[18];
  const float* fc1_b   = (const float*)d_in[19];
  const float* fc2_w   = (const float*)d_in[20];
  const float* fc2_b   = (const float*)d_in[21];
  const float* mp[8];
  for (int i = 0; i < 8; ++i) mp[i] = (const float*)d_in[22 + i];

  char* ws = (char*)d_ws;
  size_t off = 0;
  auto take = [&](size_t bytes) -> char* {
    char* p = ws + off;
    off += (bytes + 255) & ~(size_t)255;
    return p;
  };

  _Float16* emb16 = (_Float16*)take((size_t)1000 * 64 * 2);
  _Float16* wcih  = (_Float16*)take((size_t)512 * 64 * 2);
  _Float16* wchh  = (_Float16*)take((size_t)512 * 128 * 2);
  _Float16* wcfih = (_Float16*)take((size_t)1024 * 128 * 2);
  _Float16* wcfhh = (_Float16*)take((size_t)1024 * 256 * 2);
  _Float16* wcbih = (_Float16*)take((size_t)1024 * 128 * 2);
  _Float16* wcbhh = (_Float16*)take((size_t)1024 * 256 * 2);
  _Float16* wafih = (_Float16*)take((size_t)512 * 32 * 2);
  _Float16* wafhh = (_Float16*)take((size_t)512 * 128 * 2);
  _Float16* wabih = (_Float16*)take((size_t)512 * 32 * 2);
  _Float16* wabhh = (_Float16*)take((size_t)512 * 128 * 2);

  _Float16* seq16 = (_Float16*)take((size_t)8192 * 128 * 2);
  float*    ctx32 = (float*)take(4 * NCTX * 4);      // [p-fw, p-bw, h-fw, h-bw]
  _Float16* ctx16 = (_Float16*)take(4 * NCTX * 2);
  float*    S     = (float*)take((size_t)2 * 64 * 5 * 4096 * 4);
  float*    invn  = (float*)take((size_t)2 * 2 * 5 * 4096 * 4);
  float*    feats = (float*)take(8 * NCTX * 4);
  float*    mv_p  = (float*)take((size_t)64 * 64 * 32 * 4);
  float*    mv_h  = (float*)take((size_t)64 * 64 * 32 * 4);
  float*    xcat  = (float*)take((size_t)64 * 512 * 4);

  auto cvt = [&](const float* s, _Float16* d, int n) {
    f32_to_f16_kernel<<<dim3((n + 255) / 256), dim3(256), 0, stream>>>(s, d, n);
  };
  cvt(emb,    emb16, 1000 * 64);
  cvt(wih_c,  wcih,  512 * 64);
  cvt(whh_c,  wchh,  512 * 128);
  cvt(wih_cf, wcfih, 1024 * 128);
  cvt(whh_cf, wcfhh, 1024 * 256);
  cvt(wih_cb, wcbih, 1024 * 128);
  cvt(whh_cb, wcbhh, 1024 * 256);
  cvt(wih_af, wafih, 512 * 32);
  cvt(whh_af, wafhh, 512 * 128);
  cvt(wih_ab, wabih, 512 * 32);
  cvt(whh_ab, wabhh, 512 * 128);

  char_lstm_kernel<<<dim3(512), dim3(128), SMEM_CHAR, stream>>>(
      p_char, h_char, emb16, wcih, wchh, b_c, seq16);

  ctx_lstm_kernel<<<dim3(4), dim3(512), SMEM_CTX, stream>>>(
      seq16, wcfih, wcfhh, b_cf, wcbih, wcbhh, b_cb, ctx32, ctx16);

  norms_kernel<<<dim3(256), dim3(320), 0, stream>>>(ctx32, mp[2], mp[3], invn);
  pairwise_kernel<<<dim3(640), dim3(256), 0, stream>>>(ctx16, mp[2], mp[3], S);
  maxreduce_kernel<<<dim3(128), dim3(256), 0, stream>>>(S, invn, mv_p, mv_h);
  attfeat_kernel<<<dim3(16384), dim3(256), 0, stream>>>(S, invn, ctx32, feats);

  const float* pfw = ctx32 + 0 * NCTX;
  const float* pbw = ctx32 + 1 * NCTX;
  const float* hfw = ctx32 + 2 * NCTX;
  const float* hbw = ctx32 + 3 * NCTX;

  // full matches (mp1 fw, mp2 bw)
  matchfull_kernel<<<dim3(4096), dim3(128), 0, stream>>>(pfw, hfw, 63, mp[0], mv_p, 0);
  matchfull_kernel<<<dim3(4096), dim3(128), 0, stream>>>(hfw, pfw, 63, mp[0], mv_h, 0);
  matchfull_kernel<<<dim3(4096), dim3(128), 0, stream>>>(pbw, hbw, 0,  mp[1], mv_p, 16);
  matchfull_kernel<<<dim3(4096), dim3(128), 0, stream>>>(hbw, pbw, 0,  mp[1], mv_h, 16);
  // attention-mean matches (mp5 fw, mp6 bw)
  matchfull_kernel<<<dim3(4096), dim3(128), 0, stream>>>(pfw, feats + 0 * NCTX, -1, mp[4], mv_p, 8);
  matchfull_kernel<<<dim3(4096), dim3(128), 0, stream>>>(pbw, feats + 1 * NCTX, -1, mp[5], mv_p, 24);
  matchfull_kernel<<<dim3(4096), dim3(128), 0, stream>>>(hfw, feats + 2 * NCTX, -1, mp[4], mv_h, 8);
  matchfull_kernel<<<dim3(4096), dim3(128), 0, stream>>>(hbw, feats + 3 * NCTX, -1, mp[5], mv_h, 24);
  // attention-max matches (mp7 fw, mp8 bw)
  matchfull_kernel<<<dim3(4096), dim3(128), 0, stream>>>(pfw, feats + 4 * NCTX, -1, mp[6], mv_p, 12);
  matchfull_kernel<<<dim3(4096), dim3(128), 0, stream>>>(pbw, feats + 5 * NCTX, -1, mp[7], mv_p, 28);
  matchfull_kernel<<<dim3(4096), dim3(128), 0, stream>>>(hfw, feats + 6 * NCTX, -1, mp[6], mv_h, 12);
  matchfull_kernel<<<dim3(4096), dim3(128), 0, stream>>>(hbw, feats + 7 * NCTX, -1, mp[7], mv_h, 28);

  agg_lstm_kernel<<<dim3(4), dim3(256), SMEM_AGG, stream>>>(
      mv_p, mv_h, wafih, wafhh, b_af, wabih, wabhh, b_ab, xcat);

  fc_kernel<<<dim3(64), dim3(256), 0, stream>>>(xcat, fc1_w, fc1_b, fc2_w, fc2_b,
                                                (float*)d_out);
}
